// RNNModel_16054587753046
// MI455X (gfx1250) — matrix operand order repaired and checked by reading, not compiled, not run on hardware
//
#include <hip/hip_runtime.h>
#include <math.h>

// ---------------------------------------------------------------------------
// Problem constants (from reference): H=1024 hidden, V=32000 vocab, T=4096.
// ---------------------------------------------------------------------------
#define Hh 1024
#define Vv 32000
#define Tt 4096
#define M_REAL (Tt + 1)           // 4097 output rows
#define BM 64                     // GEMM block tile rows
#define BN 256                    // GEMM block tile cols
#define BK 32                     // GEMM K chunk (one wmma K)
#define M_BLOCKS 65               // ceil(4097/64)
#define M_PAD (M_BLOCKS * BM)     // 4160 (padded, zero rows beyond 4097)
#define NKCHUNK (Hh / BK)         // 32

#define NWG_RNN 16                // persistent workgroups for recurrence
#define ROWS_PER_WG (Hh / NWG_RNN) // 64 rows of W_hh per WGP (256 KB LDS)

typedef __attribute__((ext_vector_type(16))) __bf16 v16bf;
typedef __attribute__((ext_vector_type(8)))  __bf16 v8bf;
typedef __attribute__((ext_vector_type(8)))  float  v8f;
typedef __attribute__((ext_vector_type(4)))  int    v4i;

// float -> bf16 (round to nearest even), stored as raw u16
static __device__ __forceinline__ unsigned short f2bf(float x) {
    unsigned u = __float_as_uint(x);
    u += 0x7FFFu + ((u >> 16) & 1u);
    return (unsigned short)(u >> 16);
}

// ---------------------------------------------------------------------------
// Async global->LDS (CDNA5 GLOBAL_LOAD_ASYNC_TO_LDS_B128, ASYNCcnt-tracked),
// with a synchronous fallback if the builtins are not declared.
// Builtin expects (v4i AS1*, v4i AS3*, imm offset, imm cpol).
// ---------------------------------------------------------------------------
#if __has_builtin(__builtin_amdgcn_global_load_async_to_lds_b128) && \
    __has_builtin(__builtin_amdgcn_s_wait_asynccnt)
#define HAVE_ASYNC_LDS 1
#else
#define HAVE_ASYNC_LDS 0
#endif

static __device__ __forceinline__ void cp16_g2l(const unsigned short* g,
                                                unsigned short* l) {
#if HAVE_ASYNC_LDS
    __builtin_amdgcn_global_load_async_to_lds_b128(
        (__attribute__((address_space(1))) v4i*)(g),
        (__attribute__((address_space(3))) v4i*)(l), 0, 0);
#else
    *(v8bf*)l = *(const v8bf*)g;
#endif
}

static __device__ __forceinline__ void wait_async_le5() {
#if HAVE_ASYNC_LDS
    __builtin_amdgcn_s_wait_asynccnt(5);
#endif
}
static __device__ __forceinline__ void wait_async_le0() {
#if HAVE_ASYNC_LDS
    __builtin_amdgcn_s_wait_asynccnt(0);
#endif
}

// ---------------------------------------------------------------------------
// Init: h_buf[0] = initial_h (fp32), all_h_bf16 row 0 = bf16(initial_h)
// ---------------------------------------------------------------------------
__global__ void rnn_init_kernel(const float* __restrict__ h0,
                                float* __restrict__ hbuf,
                                unsigned short* __restrict__ all_h_bf) {
    int i = blockIdx.x * blockDim.x + threadIdx.x;
    if (i < Hh) {
        float v = h0[i];
        hbuf[i] = v;
        all_h_bf[i] = f2bf(v);
    }
}

// ---------------------------------------------------------------------------
// W_hy fp32 -> bf16 (one pass; W_hy then lives in L2 for the GEMM)
// ---------------------------------------------------------------------------
__global__ void cvt_bf16_kernel(const float* __restrict__ src,
                                unsigned short* __restrict__ dst, int n) {
    int i = blockIdx.x * blockDim.x + threadIdx.x;
    if (i < n) dst[i] = f2bf(src[i]);
}

// ---------------------------------------------------------------------------
// Persistent recurrence: 16 WGs x 256 threads. WG g owns rows [g*64, g*64+64)
// of W_hh, held in LDS (256 KB). Each step: load h_t (4 KB) from L2 into LDS,
// 64 dot-products (4 threads/row), tanh, publish h_{t+1} fp32 + bf16, then a
// device-scope arrive counter acts as the grid barrier (monotonic, no reset).
// ---------------------------------------------------------------------------
__global__ void __launch_bounds__(256, 1)
rnn_recurrence_kernel(const int* __restrict__ tok,
                      const float* __restrict__ W_xh,
                      const float* __restrict__ W_hh,
                      const float* __restrict__ B_h,
                      float* __restrict__ hbuf,              // 2*H fp32 double buffer
                      unsigned short* __restrict__ all_h_bf, // [M_PAD, H] bf16
                      unsigned* __restrict__ counter) {
    extern __shared__ float lds[];
    float* w    = lds;                          // ROWS_PER_WG * H
    float* sh   = lds + ROWS_PER_WG * Hh;       // H
    float* part = sh + Hh;                      // 256 partials

    const int tid = threadIdx.x;
    const int r0  = blockIdx.x * ROWS_PER_WG;

    // Stage this WG's 64 rows of W_hh into LDS (contiguous 256 KB block).
    for (int i = tid; i < ROWS_PER_WG * Hh; i += 256)
        w[i] = W_hh[(size_t)r0 * Hh + i];
    __syncthreads();

    const int row = tid >> 2;                   // 0..63
    const int q   = tid & 3;                    // quarter of K
    const int kq  = q * (Hh / 4);

    for (int t = 0; t < Tt; ++t) {
        // Fetch h_t from global (L2) into LDS.
        const float* hsrc = hbuf + (size_t)(t & 1) * Hh;
        for (int i = tid; i < Hh; i += 256) sh[i] = hsrc[i];
        __syncthreads();

        // Partial dot product: 256 MACs per thread, W and h from LDS.
        float acc = 0.f;
        const float* wr = w + row * Hh + kq;
        const float* hv = sh + kq;
        #pragma unroll 8
        for (int k = 0; k < Hh / 4; ++k) acc = fmaf(wr[k], hv[k], acc);
        part[tid] = acc;
        __syncthreads();

        if (tid < ROWS_PER_WG) {
            const int gr = r0 + tid;
            float s = part[tid * 4] + part[tid * 4 + 1] +
                      part[tid * 4 + 2] + part[tid * 4 + 3];
            // x_t[gr] = W_xh[gr, tok[t]]  (one-hot matvec == column gather)
            s += W_xh[(size_t)gr * Vv + tok[t]] + B_h[gr];
            float hn = tanhf(s);
            hbuf[(size_t)((t + 1) & 1) * Hh + gr] = hn;
            all_h_bf[(size_t)(t + 1) * Hh + gr]   = f2bf(hn);
        }
        __threadfence();        // make h_{t+1} visible device-wide
        __syncthreads();
        if (tid == 0) {
            atomicAdd(counter, 1u);
            const unsigned target = (unsigned)(NWG_RNN * (t + 1));
            while (__hip_atomic_load(counter, __ATOMIC_ACQUIRE,
                                     __HIP_MEMORY_SCOPE_AGENT) < target)
                __builtin_amdgcn_s_sleep(1);
        }
        __syncthreads();
    }
}

// ---------------------------------------------------------------------------
// Output projection GEMM: out[m, n] = sum_k all_h[m,k] * W_hy[n,k].
// Block = 256 threads (8 waves), block tile 64x256, K chunked by 32 and
// double-buffered through LDS via async global->LDS copies (ASYNCcnt).
// Wave grid 2x4: each wave computes 2 M-tiles x 4 N-tiles (8 accumulators)
// with v_wmma_f32_16x16x32_bf16; operands read from LDS (ds_load_b128).
// Fragment layouts per CDNA5 ISA 7.12.2 (wave32).
// ---------------------------------------------------------------------------
__global__ void __launch_bounds__(256)
out_gemm_kernel(const unsigned short* __restrict__ A,   // [M_PAD, H] bf16
                const unsigned short* __restrict__ Bm,  // [V, H] bf16 (W_hy)
                float* __restrict__ out) {              // [M_REAL, V] fp32
    __shared__ unsigned short sA[2][BM * BK];   // 2 x 4 KB
    __shared__ unsigned short sB[2][BN * BK];   // 2 x 16 KB

    const int tid  = threadIdx.x;
    const int lane = tid & 31;
    const int wave = tid >> 5;
    const int wr   = wave >> 2;      // 0..1  wave row
    const int wc   = wave & 3;       // 0..3  wave col
    const int half = lane >> 4;      // 0: lanes 0-15, 1: lanes 16-31
    const int l16  = lane & 15;

    const int mBlock = blockIdx.y * BM;
    const int nBlock = blockIdx.x * BN;

    // Per-thread staging sources: 1 x 16B for A, 4 x 16B for B per chunk.
    const unsigned short* aSrc = A + (size_t)(mBlock + (tid >> 2)) * Hh + (tid & 3) * 8;
    const unsigned short* bSrc[4];
    #pragma unroll
    for (int j = 0; j < 4; ++j)
        bSrc[j] = Bm + (size_t)(nBlock + j * 64 + (tid >> 2)) * Hh + (tid & 3) * 8;

    auto stage = [&](int buf, int k0) {
        cp16_g2l(aSrc + k0, &sA[buf][tid * 8]);
        #pragma unroll
        for (int j = 0; j < 4; ++j)
            cp16_g2l(bSrc[j] + k0, &sB[buf][(j * 256 + tid) * 8]);
    };

    const v8f zero = {};
    v8f acc[2][4];
    #pragma unroll
    for (int mt = 0; mt < 2; ++mt)
        #pragma unroll
        for (int nt = 0; nt < 4; ++nt) acc[mt][nt] = zero;

    stage(0, 0);

    for (int kc = 0; kc < NKCHUNK; ++kc) {
        const int buf = kc & 1;
        if (kc + 1 < NKCHUNK) {
            stage(buf ^ 1, (kc + 1) * BK);
            wait_async_le5();           // this wave's chunk kc is in LDS
        } else {
            wait_async_le0();
        }
        __syncthreads();                // all waves' slices of chunk kc ready

        // A fragments: lane holds row; half 0 -> K {0..7,16..23}, 1 -> {8..15,24..31}
        v16bf af[2];
        #pragma unroll
        for (int mt = 0; mt < 2; ++mt) {
            const unsigned short* p =
                &sA[buf][(wr * 32 + mt * 16 + l16) * BK + (half ? 8 : 0)];
            v8bf lo = *(const v8bf*)p;
            v8bf hi = *(const v8bf*)(p + 16);
            af[mt] = __builtin_shufflevector(lo, hi, 0, 1, 2, 3, 4, 5, 6, 7,
                                             8, 9, 10, 11, 12, 13, 14, 15);
        }
        // B fragments: lane holds column; half 0 -> K {0..7,16..23}, 1 -> {8..15,24..31}
        v16bf bfr[4];
        #pragma unroll
        for (int nt = 0; nt < 4; ++nt) {
            const unsigned short* p =
                &sB[buf][(wc * 64 + nt * 16 + l16) * BK + (half ? 8 : 0)];
            v8bf lo = *(const v8bf*)p;
            v8bf hi = *(const v8bf*)(p + 16);
            bfr[nt] = __builtin_shufflevector(lo, hi, 0, 1, 2, 3, 4, 5, 6, 7,
                                              8, 9, 10, 11, 12, 13, 14, 15);
        }

        #pragma unroll
        for (int mt = 0; mt < 2; ++mt)
            #pragma unroll
            for (int nt = 0; nt < 4; ++nt)
                acc[mt][nt] = __builtin_amdgcn_wmma_f32_16x16x32_bf16(
                    false, af[mt], false, bfr[nt], (short)0, acc[mt][nt],
                    false, false);

        __syncthreads();                // done reading buf before it's rewritten
    }

    // D layout: VGPR r -> row +r (lanes 0-15) / +8+r (lanes 16-31), col = l16
    #pragma unroll
    for (int mt = 0; mt < 2; ++mt)
        #pragma unroll
        for (int nt = 0; nt < 4; ++nt) {
            const int n = nBlock + wc * 64 + nt * 16 + l16;
            #pragma unroll
            for (int r = 0; r < 8; ++r) {
                const int m = mBlock + wr * 32 + mt * 16 + r + half * 8;
                if (m < M_REAL) out[(size_t)m * Vv + n] = acc[mt][nt][r];
            }
        }
}

// ---------------------------------------------------------------------------
// Host side
// ---------------------------------------------------------------------------
extern "C" void kernel_launch(void* const* d_in, const int* in_sizes, int n_in,
                              void* d_out, int out_size, void* d_ws, size_t ws_size,
                              hipStream_t stream) {
    const int*   tok   = (const int*)d_in[0];
    const float* h0    = (const float*)d_in[1];
    const float* W_xh  = (const float*)d_in[2];
    const float* W_hh  = (const float*)d_in[3];
    const float* W_hy  = (const float*)d_in[4];
    const float* B_h   = (const float*)d_in[5];
    float*       out   = (float*)d_out;

    // Workspace layout
    char* ws = (char*)d_ws;
    unsigned*       counter = (unsigned*)ws;                         // [0, 64)
    float*          hbuf    = (float*)(ws + 64);                     // 2*H fp32
    unsigned short* all_h   = (unsigned short*)(ws + 8448);          // M_PAD*H bf16
    unsigned short* why_bf  = (unsigned short*)(ws + 8448 + (size_t)M_PAD * Hh * 2);

    // Zero the barrier counter and the padded all_h (rows >= 4097 stay zero).
    (void)hipMemsetAsync(counter, 0, 64, stream);
    (void)hipMemsetAsync(all_h, 0, (size_t)M_PAD * Hh * 2, stream);

    rnn_init_kernel<<<dim3((Hh + 255) / 256), dim3(256), 0, stream>>>(h0, hbuf, all_h);

    const int nWhy = Vv * Hh;
    cvt_bf16_kernel<<<dim3((nWhy + 255) / 256), dim3(256), 0, stream>>>(W_hy, why_bf, nWhy);

    const size_t ldsBytes = (size_t)(ROWS_PER_WG * Hh + Hh + 256) * sizeof(float);
    rnn_recurrence_kernel<<<dim3(NWG_RNN), dim3(256), ldsBytes, stream>>>(
        tok, W_xh, W_hh, B_h, hbuf, all_h, counter);

    out_gemm_kernel<<<dim3(Vv / BN, M_BLOCKS), dim3(256), 0, stream>>>(all_h, why_bf, out);
}
